// DeformNet_12867722019588
// MI455X (gfx1250) — matrix operand order repaired and checked by reading, not compiled, not run on hardware
//
#include <hip/hip_runtime.h>
#include <hip/hip_bf16.h>
#include <math.h>

// DeformNet fused: hash-grid encode (14 levels, trilinear+smoothstep, spatial hash)
// + 36->64->64->3 tanh MLP via v_wmma_f32_16x16x32_f16, residual in bbox space.
// One wave32 = one 16-point tile. Features staged in LDS, weights pre-swizzled
// (N-major f16) in LDS so WMMA A/B fragments are contiguous 16B ds loads.
// tanh is branchless (native v_tanh_f32 if available, else exp2+rcp).

typedef __attribute__((ext_vector_type(8)))  _Float16 v8h;
typedef __attribute__((ext_vector_type(16))) _Float16 v16h;
typedef __attribute__((ext_vector_type(8)))  float    v8f;

#define LDS_WAIT() asm volatile("s_wait_dscnt 0" ::: "memory")

// floor(16 * 1.32^l), l = 0..13
__constant__ float kResF[14] = {16.f, 21.f, 27.f, 36.f, 48.f, 64.f, 84.f,
                                111.f, 147.f, 194.f, 256.f, 339.f, 447.f, 590.f};

__device__ __forceinline__ float fast_tanh(float v) {
#if defined(__HIP_DEVICE_COMPILE__)
#if __has_builtin(__builtin_amdgcn_tanhf)
  return __builtin_amdgcn_tanhf(v);                 // native v_tanh_f32 (TRANS32)
#else
  // Branchless: tanh(v) = 1 - 2/(2^(2*log2e*v) + 1); exact at +/-1 saturation.
  const float t = __builtin_amdgcn_exp2f(v * 2.8853900817779268f); // v_exp_f32
  return 1.0f - 2.0f * __builtin_amdgcn_rcpf(t + 1.0f);            // v_rcp_f32
#endif
#else
  return tanhf(v);  // host pass only; never executed on device
#endif
}

__device__ __forceinline__ float fast_rcp(float v) {
#if defined(__HIP_DEVICE_COMPILE__)
  return __builtin_amdgcn_rcpf(v);
#else
  return 1.0f / v;
#endif
}

__device__ __forceinline__ v16h lds2x8(const _Float16* p, int offA, int offB) {
  v8h a = *(const v8h*)(p + offA);   // 16B ds_load_b128
  v8h b = *(const v8h*)(p + offB);   // 16B ds_load_b128
  return __builtin_shufflevector(a, b, 0, 1, 2, 3, 4, 5, 6, 7,
                                       8, 9, 10, 11, 12, 13, 14, 15);
}

__device__ __forceinline__ v8f wmma16(v16h a, v16h b, v8f c) {
  // (neg_a, A, neg_b, B, c_mod, C, reuse_a, reuse_b)
  return __builtin_amdgcn_wmma_f32_16x16x32_f16(false, a, false, b,
                                                (short)0, c, false, false);
}

__global__ __launch_bounds__(256) void deformnet_fused(
    const float* __restrict__ x, const float* __restrict__ e,
    const float* __restrict__ tables,
    const float* __restrict__ W1, const float* __restrict__ B1,
    const float* __restrict__ W2, const float* __restrict__ B2,
    const float* __restrict__ W3, const float* __restrict__ B3,
    const float* __restrict__ bbox, float* __restrict__ out, int n) {
  // Weights pre-swizzled to N-major (column-contiguous) f16; K padded to 64.
  __shared__ __align__(32) _Float16 w1t[64 * 64];   // [n][k], k>=36 -> 0
  __shared__ __align__(32) _Float16 w2t[64 * 64];   // [n][k]
  __shared__ __align__(32) _Float16 w3t[16 * 64];   // [n][k], n>=3 -> 0
  __shared__ float b1l[64], b2l[64], b3l[16];
  __shared__ __align__(32) _Float16 stage[8][16 * 64];  // per-wave 16x64 tile

  const int tid = threadIdx.x;

  // ---- cooperative weight load + transpose + f32->f16 ----
  for (int idx = tid; idx < 64 * 64; idx += 256) {
    int nn = idx >> 6, kk = idx & 63;
    w1t[idx] = (kk < 36) ? (_Float16)W1[kk * 64 + nn] : (_Float16)0.f;
    w2t[idx] = (_Float16)W2[kk * 64 + nn];
  }
  for (int idx = tid; idx < 16 * 64; idx += 256) {
    int nn = idx >> 6, kk = idx & 63;
    w3t[idx] = (nn < 3) ? (_Float16)W3[kk * 3 + nn] : (_Float16)0.f;
  }
  if (tid < 64) { b1l[tid] = B1[tid]; b2l[tid] = B2[tid]; }
  if (tid < 16) b3l[tid] = (tid < 3) ? B3[tid] : 0.f;
  __syncthreads();

  const int wv   = tid >> 5;
  const int lane = tid & 31;
  const int tile = blockIdx.x * 8 + wv;
  const int row0 = tile * 16;
  _Float16* stg  = &stage[wv][0];

  const float lox = bbox[0], loy = bbox[1], loz = bbox[2];
  const float ivx = fast_rcp(bbox[3] - lox);
  const float ivy = fast_rcp(bbox[4] - loy);
  const float ivz = fast_rcp(bbox[5] - loz);

  // ---------------- hash-grid encode: 2 lanes/point, 7 levels/lane ----------
  {
    const int p    = lane >> 1;
    const int oddl = lane & 1;
    const int row  = row0 + p;
    const int lb   = oddl * 7;
    _Float16* srow = stg + p * 64;
    if (row < n) {
      const float xr = x[row * 3 + 0], yr = x[row * 3 + 1], zr = x[row * 3 + 2];
      const float xnx = (xr - lox) * ivx;
      const float xny = (yr - loy) * ivy;
      const float xnz = (zr - loz) * ivz;
      const float2* t2 = (const float2*)tables;
      for (int i = 0; i < 7; ++i) {
        const int lvl = lb + i;
        const float res = kResF[lvl];
        const float px = xnx * res, py = xny * res, pz = xnz * res;
        const float bxf = floorf(px), byf = floorf(py), bzf = floorf(pz);
        const float fx = px - bxf, fy = py - byf, fz = pz - bzf;
        const float wx = fx * fx * (3.f - 2.f * fx);
        const float wy = fy * fy * (3.f - 2.f * fy);
        const float wz = fz * fz * (3.f - 2.f * fz);
        const uint32_t bx = (uint32_t)(int)bxf;
        const uint32_t by = (uint32_t)(int)byf;
        const uint32_t bz = (uint32_t)(int)bzf;
        const uint32_t tb = ((uint32_t)lvl) << 19;
        float f0 = 0.f, f1 = 0.f;
#pragma unroll
        for (int c = 0; c < 8; ++c) {
          uint32_t h = (bx + (c & 1))
                     ^ ((by + ((c >> 1) & 1)) * 2654435761u)
                     ^ ((bz + ((c >> 2) & 1)) * 805459861u);
          h &= 0x7FFFFu;
          const float2 t = t2[tb + h];                  // global_load_b64 gather (L2)
          const float wgt = ((c & 1)        ? wx : 1.f - wx)
                          * (((c >> 1) & 1) ? wy : 1.f - wy)
                          * (((c >> 2) & 1) ? wz : 1.f - wz);
          f0 = fmaf(wgt, t.x, f0);
          f1 = fmaf(wgt, t.y, f1);
        }
        srow[2 * lvl + 0] = (_Float16)f0;
        srow[2 * lvl + 1] = (_Float16)f1;
      }
      if (!oddl) {
#pragma unroll
        for (int j = 0; j < 8; ++j) srow[28 + j] = (_Float16)e[row * 8 + j];
      } else {
#pragma unroll
        for (int j = 0; j < 28; ++j) srow[36 + j] = (_Float16)0.f;
      }
    } else {
      for (int j = 0; j < 14; ++j) srow[2 * lb + j] = (_Float16)0.f;
      if (!oddl) for (int j = 0; j < 8;  ++j) srow[28 + j] = (_Float16)0.f;
      else       for (int j = 0; j < 28; ++j) srow[36 + j] = (_Float16)0.f;
    }
  }
  LDS_WAIT();

  // ---------------- MLP via WMMA ----------------
  const int r  = lane & 15;   // row-in-tile for A, col for B/C
  const int hi = lane >> 4;
  const _Float16* rowp = stg + r * 64;

  // layers 1 & 2: 64-wide, tanh, restage as f16
  for (int layer = 0; layer < 2; ++layer) {
    const _Float16* wt = layer ? w2t : w1t;
    const float*    bl = layer ? b2l : b1l;
    // A fragments: K chunks 0-31 and 32-63 (16-bit A layout)
    const v16h a0 = lds2x8(rowp,      hi * 8, 16 + hi * 8);
    const v16h a1 = lds2x8(rowp, 32 + hi * 8, 48 + hi * 8);
#pragma unroll
    for (int nb = 0; nb < 4; ++nb) {
      const _Float16* cb = wt + (nb * 16 + r) * 64;          // column n, contiguous K
      const v16h b0 = lds2x8(cb,      hi * 8, 16 + hi * 8);
      const v16h b1 = lds2x8(cb, 32 + hi * 8, 48 + hi * 8);
      v8f c = {0.f, 0.f, 0.f, 0.f, 0.f, 0.f, 0.f, 0.f};
      c = wmma16(a0, b0, c);
      c = wmma16(a1, b1, c);
      const float bb = bl[nb * 16 + r];
#pragma unroll
      for (int v = 0; v < 8; ++v) {
        const float y = fast_tanh(c[v] + bb);
        stg[(hi * 8 + v) * 64 + nb * 16 + r] = (_Float16)y;  // C layout -> restage
      }
    }
    LDS_WAIT();
  }

  // layer 3: 64 -> 3 (padded to 16 cols), residual + bbox rescale
  {
    const v16h a0 = lds2x8(rowp,      hi * 8, 16 + hi * 8);
    const v16h a1 = lds2x8(rowp, 32 + hi * 8, 48 + hi * 8);
    const _Float16* cb = w3t + r * 64;
    const v16h b0 = lds2x8(cb,      hi * 8, 16 + hi * 8);
    const v16h b1 = lds2x8(cb, 32 + hi * 8, 48 + hi * 8);
    v8f c = {0.f, 0.f, 0.f, 0.f, 0.f, 0.f, 0.f, 0.f};
    c = wmma16(a0, b0, c);
    c = wmma16(a1, b1, c);
    if (r < 3) {
      const float scale = bbox[3 + r] - bbox[r];
      const float bb = b3l[r];
#pragma unroll
      for (int v = 0; v < 8; ++v) {
        const int row = row0 + hi * 8 + v;
        if (row < n) {
          // (mlp + xn)*(hi-lo) + lo  ==  mlp*(hi-lo) + x
          out[row * 3 + r] = (c[v] + bb) * scale + x[row * 3 + r];
        }
      }
    }
  }
}

extern "C" void kernel_launch(void* const* d_in, const int* in_sizes, int n_in,
                              void* d_out, int out_size, void* d_ws, size_t ws_size,
                              hipStream_t stream) {
  (void)n_in; (void)out_size; (void)d_ws; (void)ws_size;
  const float* x      = (const float*)d_in[0];
  const float* e      = (const float*)d_in[1];
  const float* tables = (const float*)d_in[2];
  const float* W1     = (const float*)d_in[3];
  const float* B1     = (const float*)d_in[4];
  const float* W2     = (const float*)d_in[5];
  const float* B2     = (const float*)d_in[6];
  const float* W3     = (const float*)d_in[7];
  const float* B3     = (const float*)d_in[8];
  const float* bbox   = (const float*)d_in[9];
  float* out = (float*)d_out;

  const int n = in_sizes[0] / 3;
  if (n <= 0) return;
  const int tiles  = (n + 15) / 16;        // one wave per 16-point tile
  const int blocks = (tiles + 7) / 8;      // 8 waves per 256-thread block
  deformnet_fused<<<blocks, 256, 0, stream>>>(x, e, tables, W1, B1, W2, B2,
                                              W3, B3, bbox, out, n);
}
